// NeuroSparseTransformerBlock_23338852286586
// MI455X (gfx1250) — matrix operand order repaired and checked by reading, not compiled, not run on hardware
//
#include <hip/hip_runtime.h>
#include <hip/hip_bf16.h>

// ---------------- problem constants ----------------
constexpr int BB   = 4;
constexpr int LV   = 1024;
constexpr int LA   = 512;
constexpr int LI   = 256;
constexpr int T    = LV + LA + LI;      // 1792
constexpr int TOPK = 537;               // int(0.3 * T)
constexpr int KPAD = 544;               // 34 * 16
constexpr int D    = 1024;
constexpr int H    = 16;
constexpr int DH   = 64;
constexpr int FF   = 4096;
constexpr int MTILES = KPAD / 16;       // 34
constexpr float DECAY = 0.9f;

// GEMM tiling
constexpr int BM  = 64;
constexpr int BN  = 64;
constexpr int BKT = 64;

typedef __attribute__((ext_vector_type(16))) __bf16   v16bf;
typedef __attribute__((ext_vector_type(8)))  float    v8f;
typedef __attribute__((ext_vector_type(4)))  unsigned uv4;
typedef __attribute__((ext_vector_type(8)))  int      iv8;
typedef __attribute__((ext_vector_type(4)))  int      iv4;

// ---------------- small helpers ----------------
__device__ inline __bf16 f2bf(float f) {
  union { float f; unsigned u; } a; a.f = f;
  unsigned r = a.u + 0x7FFFu + ((a.u >> 16) & 1u);   // round-to-nearest-even
  union { unsigned short s; __bf16 b; } c; c.s = (unsigned short)(r >> 16);
  return c.b;
}

__device__ inline float block_reduce_sum(float v, float* buf) {
  int tid = threadIdx.x;
  buf[tid] = v;
  __syncthreads();
  for (int s = blockDim.x >> 1; s > 0; s >>= 1) {
    if (tid < s) buf[tid] += buf[tid + s];
    __syncthreads();
  }
  float r = buf[0];
  __syncthreads();
  return r;
}

// ---------------- WMMA fragment loaders (contiguous k order: k = 16*(lane/16) + elem) ----------------
// A fragment (16x32 bf16) from row-major source: lanes 0-15 (M=lane) hold
// elems 0-15 -> K k0..k0+15; lanes 16-31 hold K k0+16..k0+31.
__device__ inline v16bf load_a_rm(const __bf16* A, int lda, int mBase, int k0, int lane) {
  const __bf16* p = A + (size_t)(mBase + (lane & 15)) * lda + k0 + ((lane & 16) ? 16 : 0);
  v16bf a;
#pragma unroll
  for (int e = 0; e < 8; ++e) a[e] = p[e];
#pragma unroll
  for (int e = 0; e < 8; ++e) a[e + 8] = p[e + 8];
  return a;
}

// Vectorized A fragment from LDS (16B-aligned rows).
__device__ inline v16bf lds_load_a(const __bf16* As, int lda, int mOff, int kk, int lane) {
  const __bf16* p = As + (size_t)(mOff + (lane & 15)) * lda + kk + ((lane & 16) ? 16 : 0);
  union { uv4 q; __bf16 h[8]; } u0, u1;
  u0.q = *(const uv4*)p;
  u1.q = *(const uv4*)(p + 8);
  v16bf a;
#pragma unroll
  for (int e = 0; e < 8; ++e) { a[e] = u0.h[e]; a[e + 8] = u1.h[e]; }
  return a;
}

// B fragment (32x16) from row-major (Kd x N): lane n = nBase+(lane&15);
// lanes 0-15 hold K k0..k0+15, lanes 16-31 hold K k0+16..k0+31.
__device__ inline v16bf load_b_rm(const __bf16* W, int ldw, int k0, int nBase, int lane) {
  int n  = nBase + (lane & 15);
  int kk = k0 + ((lane & 16) ? 16 : 0);
  v16bf b;
#pragma unroll
  for (int e = 0; e < 16; ++e) b[e] = W[(size_t)(kk + e) * ldw + n];
  return b;
}

// B fragment where effective B[kd][n] = S[n][kd], S row-major (N x Kd).
__device__ inline v16bf load_bt_rm(const __bf16* S, int lds_, int k0, int nBase, int lane) {
  const __bf16* p = S + (size_t)(nBase + (lane & 15)) * lds_ + k0 + ((lane & 16) ? 16 : 0);
  v16bf b;
#pragma unroll
  for (int e = 0; e < 16; ++e) b[e] = p[e];
  return b;
}

__device__ inline v8f wmma_bf16(v16bf a, v16bf b, v8f c) {
  return __builtin_amdgcn_wmma_f32_16x16x32_bf16(false, a, false, b, (short)0, c, false, false);
}

// ---------------- TDM 2-D tile load (D# per ISA cdna5 §8) ----------------
#if __has_builtin(__builtin_amdgcn_tensor_load_to_lds)
#define HAVE_TDM 1
__device__ inline void tdm_load_2d(unsigned lds_byte_off, const void* gaddr,
                                   int tile_x, int tile_y, long row_stride_elems) {
  unsigned long long ga = (unsigned long long)(size_t)gaddr;
  uv4 g0;
  g0[0] = 1u;                                            // count=1, user descriptor
  g0[1] = lds_byte_off;                                  // lds_addr
  g0[2] = (unsigned)(ga & 0xFFFFFFFFu);                  // global_addr[31:0]
  g0[3] = (unsigned)((ga >> 32) & 0x01FFFFFFu) | (2u << 30); // addr[56:32] | type=2
  unsigned td0 = (unsigned)tile_x;                       // tensor dims = tile (no OOB)
  unsigned td1 = (unsigned)tile_y;
  unsigned long long s0 = (unsigned long long)row_stride_elems;
  iv8 g1;
  g1[0] = (int)(1u << 16);                               // wg_mask=0, data_size=1 (2B)
  g1[1] = (int)((td0 & 0xFFFFu) << 16);                  // ab_addr=0 | td0[15:0]
  g1[2] = (int)(((td0 >> 16) & 0xFFFFu) | ((td1 & 0xFFFFu) << 16));
  g1[3] = (int)(((td1 >> 16) & 0xFFFFu) | ((unsigned)tile_x << 16)); // tile_dim0
  g1[4] = (int)((unsigned)tile_y & 0xFFFFu);             // tile_dim1 | tile_dim2=0
  g1[5] = (int)(s0 & 0xFFFFFFFFu);                       // dim0 stride lo
  g1[6] = (int)((s0 >> 32) & 0xFFFFu);                   // dim0 stride hi | dim1 stride=0
  g1[7] = 0;
  iv4 gz4 = {0, 0, 0, 0};
  iv8 gz8 = {0, 0, 0, 0, 0, 0, 0, 0};
  // 6-arg form (clang-23 / therock-10.0): (g0, g1, g2, g3, g4, cpol)
  __builtin_amdgcn_tensor_load_to_lds(g0, g1, gz4, gz4, gz8, 0);
}
#else
#define HAVE_TDM 0
#endif

// ---------------- kernels ----------------

__global__ void cvt_bf16_kernel(const float* __restrict__ in, __bf16* __restrict__ out, long n) {
  long i = (long)blockIdx.x * blockDim.x + threadIdx.x;
  if (i < n) out[i] = f2bf(in[i]);
}

// LN1 of one token row + current = y . w_cur, membrane update.
__global__ void ln1_membrane_kernel(const float* __restrict__ vis, const float* __restrict__ aud,
                                    const float* __restrict__ imu, const float* __restrict__ g,
                                    const float* __restrict__ beta, const float* __restrict__ wcur,
                                    const float* __restrict__ bcur, const float* __restrict__ memb,
                                    __bf16* __restrict__ xn, float* __restrict__ newmem) {
  __shared__ float red[256];
  int r = blockIdx.x;
  int b = r / T, t = r % T;
  const float* src = (t < LV)      ? vis + ((size_t)b * LV + t) * D
                   : (t < LV + LA) ? aud + ((size_t)b * LA + (t - LV)) * D
                                   : imu + ((size_t)b * LI + (t - LV - LA)) * D;
  float vals[4];
  float s = 0.f, sq = 0.f;
#pragma unroll
  for (int i = 0; i < 4; ++i) {
    int d = threadIdx.x + i * 256;
    float x = src[d];
    vals[i] = x; s += x; sq += x * x;
  }
  s  = block_reduce_sum(s, red);
  sq = block_reduce_sum(sq, red);
  float m   = s * (1.0f / D);
  float var = sq * (1.0f / D) - m * m;
  float rs  = rsqrtf(var + 1e-5f);
  float cur = 0.f;
#pragma unroll
  for (int i = 0; i < 4; ++i) {
    int d = threadIdx.x + i * 256;
    float y = (vals[i] - m) * rs * g[d] + beta[d];
    xn[(size_t)r * D + d] = f2bf(y);
    cur += y * wcur[d];
  }
  cur = block_reduce_sum(cur, red);
  if (threadIdx.x == 0) newmem[r] = DECAY * memb[r] + cur + bcur[0];
}

// Per-batch top-K via bitonic sort (value desc, index asc tie order).
__global__ void topk_kernel(const float* __restrict__ newmem, int* __restrict__ idx,
                            int* __restrict__ selpos) {
  __shared__ unsigned long long keys[2048];
  __shared__ int sidx[1024];
  int b = blockIdx.x, tid = threadIdx.x;   // 1024 threads

  for (int t = tid; t < T; t += 1024) selpos[b * T + t] = -1;

  for (int t = tid; t < 2048; t += 1024) {
    unsigned long long key;
    if (t < T) {
      float v = newmem[b * T + t];
      unsigned u = __float_as_uint(v);
      unsigned mono = (u & 0x80000000u) ? ~u : (u | 0x80000000u);
      key = ((unsigned long long)(~mono) << 32) | (unsigned)t;
    } else {
      key = 0xFFFFFFFFFFFFFFFFull;
    }
    keys[t] = key;
  }
  __syncthreads();

  for (int k = 2; k <= 2048; k <<= 1) {
    for (int j = k >> 1; j > 0; j >>= 1) {
      for (int base = tid; base < 2048; base += 1024) {
        int ixj = base ^ j;
        if (ixj > base) {
          bool up = ((base & k) == 0);
          unsigned long long a = keys[base], c = keys[ixj];
          if (up ? (a > c) : (a < c)) { keys[base] = c; keys[ixj] = a; }
        }
      }
      __syncthreads();
    }
  }

  sidx[tid] = (tid < TOPK) ? (int)(keys[tid] & 0xFFFFFFFFull) : 0x7FFFFFFF;
  __syncthreads();

  for (int k = 2; k <= 1024; k <<= 1) {
    for (int j = k >> 1; j > 0; j >>= 1) {
      int ixj = tid ^ j;
      if (ixj > tid) {
        bool up = ((tid & k) == 0);
        int a = sidx[tid], c = sidx[ixj];
        if (up ? (a > c) : (a < c)) { sidx[tid] = c; sidx[ixj] = a; }
      }
      __syncthreads();
    }
  }

  if (tid < KPAD) idx[b * KPAD + tid] = (tid < TOPK) ? sidx[tid] : -1;
  __syncthreads();
  if (tid < TOPK) selpos[b * T + sidx[tid]] = tid;
}

// Gather selected (normalized) tokens, zero-pad rows K..KPAD-1.
__global__ void gather_kernel(const __bf16* __restrict__ xn, const int* __restrict__ idx,
                              __bf16* __restrict__ xsel) {
  int row = blockIdx.x;                  // B*KPAD rows
  int b = row / KPAD, kk = row % KPAD;
  int src = (kk < TOPK) ? idx[b * KPAD + kk] : -1;
  __bf16* dp = xsel + (size_t)row * D;
  if (src >= 0) {
    const __bf16* sp = xn + ((size_t)(b * T + src)) * D;
    for (int d = threadIdx.x; d < D; d += blockDim.x) dp[d] = sp[d];
  } else {
    __bf16 z = f2bf(0.f);
    for (int d = threadIdx.x; d < D; d += blockDim.x) dp[d] = z;
  }
}

// ---------- block-tiled WMMA GEMM, double-buffered TDM pipeline ----------
// MODE 0: QKV -> bf16 out, head-major remap to (B*H, KPAD, DH)
// MODE 1: f32 out + bias (O-projection)
// MODE 2: bf16 out + bias + exact GELU (FFN1)
// MODE 3: f32 + bias + residual, split-write to vis/aud/imu (FFN2)
template <int MODE>
__global__ void gemm_tiled_kernel(const __bf16* __restrict__ A, const __bf16* __restrict__ W,
                                  const float* __restrict__ bias, void* __restrict__ out0,
                                  float* __restrict__ out1, float* __restrict__ out2,
                                  float* __restrict__ out3, const float* __restrict__ resid,
                                  int M, int N, int Kd) {
  __shared__ __align__(16) __bf16 As[2][BM * BKT];   // 2 x 8 KB
  __shared__ __align__(16) __bf16 Bs[2][BKT * BN];   // 2 x 8 KB
  int ntn = N / BN;
  int mt = blockIdx.x / ntn, nt = blockIdx.x % ntn;
  int mBase = mt * BM, nBase = nt * BN;
  int tid = threadIdx.x, lane = tid & 31, w = tid >> 5;
  int mi = w & 3;                 // 4 m-subtiles
  int ni0 = (w >> 2) * 2;         // 2 n-subtiles per wave

  v8f acc0 = {0.f, 0.f, 0.f, 0.f, 0.f, 0.f, 0.f, 0.f};
  v8f acc1 = {0.f, 0.f, 0.f, 0.f, 0.f, 0.f, 0.f, 0.f};

  const int ktiles = Kd / BKT;

#if HAVE_TDM
  unsigned ldsA[2] = {(unsigned)(size_t)&As[0][0], (unsigned)(size_t)&As[1][0]};
  unsigned ldsB[2] = {(unsigned)(size_t)&Bs[0][0], (unsigned)(size_t)&Bs[1][0]};
  // prologue: DMA tile 0 into buffer 0
  if (w == 0) {
    tdm_load_2d(ldsA[0], A + (size_t)mBase * Kd, BKT, BM, (long)Kd);
    tdm_load_2d(ldsB[0], W + nBase, BN, BKT, (long)N);
  }
#endif

  for (int it = 0; it < ktiles; ++it) {
    int cur = it & 1;
#if HAVE_TDM
    if (w == 0) {
      if (it + 1 < ktiles) {
        // prefetch tile it+1 into the alternate buffer (WAR covered by the
        // trailing block barrier of iteration it-1)
        int kn = (it + 1) * BKT;
        tdm_load_2d(ldsA[cur ^ 1], A + (size_t)mBase * Kd + kn, BKT, BM, (long)Kd);
        tdm_load_2d(ldsB[cur ^ 1], W + (size_t)kn * N + nBase, BN, BKT, (long)N);
        // TDM ops complete in order per wave: <=2 outstanding => tile `it` done
        __builtin_amdgcn_s_wait_tensorcnt(2);
      } else {
        __builtin_amdgcn_s_wait_tensorcnt(0);
      }
    }
    __syncthreads();   // tile `it` visible to all waves
#else
    cur = 0;
    __syncthreads();
    {
      int k0 = it * BKT;
      for (int i = tid; i < BM * BKT / 8; i += 256) {
        int rr = (i * 8) / BKT, cc = (i * 8) % BKT;
        *(uv4*)&As[0][i * 8] = *(const uv4*)&A[(size_t)(mBase + rr) * Kd + k0 + cc];
      }
      for (int i = tid; i < BKT * BN / 8; i += 256) {
        int rr = (i * 8) / BN, cc = (i * 8) % BN;
        *(uv4*)&Bs[0][i * 8] = *(const uv4*)&W[(size_t)(k0 + rr) * N + nBase + cc];
      }
    }
    __syncthreads();
#endif
#pragma unroll
    for (int kk = 0; kk < BKT; kk += 32) {
      v16bf a  = lds_load_a(As[cur], BKT, mi * 16, kk, lane);
      v16bf b0 = load_b_rm(Bs[cur], BN, kk, ni0 * 16, lane);
      v16bf b1 = load_b_rm(Bs[cur], BN, kk, (ni0 + 1) * 16, lane);
      acc0 = wmma_bf16(a, b0, acc0);
      acc1 = wmma_bf16(a, b1, acc1);
    }
    __syncthreads();   // reads done before this buffer is overwritten
  }

  // epilogue
#pragma unroll
  for (int half = 0; half < 2; ++half) {
    v8f acc = half ? acc1 : acc0;
    int col = nBase + (ni0 + half) * 16 + (lane & 15);
    float bv = bias[col];
#pragma unroll
    for (int r = 0; r < 8; ++r) {
      int row = mBase + mi * 16 + r + ((lane & 16) ? 8 : 0);
      float x = acc[r] + bv;
      if constexpr (MODE == 0) {
        int b = row / KPAD, kk = row % KPAD;
        int h = col >> 6, dh = col & 63;
        ((__bf16*)out0)[((size_t)(b * H + h) * KPAD + kk) * DH + dh] = f2bf(x);
      } else if constexpr (MODE == 1) {
        out1[(size_t)row * N + col] = x;
      } else if constexpr (MODE == 2) {
        float ge = 0.5f * x * (1.0f + erff(x * 0.70710678118654752f));
        ((__bf16*)out0)[(size_t)row * N + col] = f2bf(ge);
      } else {
        float val = x + resid[(size_t)row * D + col];
        int b = row / T, t = row % T;
        float* dst = (t < LV)      ? out1 + ((size_t)b * LV + t) * D
                   : (t < LV + LA) ? out2 + ((size_t)b * LA + (t - LV)) * D
                                   : out3 + ((size_t)b * LI + (t - LV - LA)) * D;
        dst[col] = val;
      }
    }
  }
}

// Flash-style attention: one block per (b*H+h, 16-row tile). 128 threads.
__global__ void attn_kernel(const __bf16* __restrict__ q, const __bf16* __restrict__ k,
                            const __bf16* __restrict__ v, __bf16* __restrict__ ctx) {
  __shared__ float  sc[16][KPAD];
  __shared__ __bf16 at[16][KPAD];
  int bid   = blockIdx.x;
  int mtile = bid % MTILES;
  int bh    = bid / MTILES;
  const __bf16* Q  = q + (size_t)bh * KPAD * DH;
  const __bf16* Km = k + (size_t)bh * KPAD * DH;
  const __bf16* V  = v + (size_t)bh * KPAD * DH;
  int lane = threadIdx.x & 31, wid = threadIdx.x >> 5;
  int mBase = mtile * 16;

  for (int nt = wid; nt < MTILES; nt += 4) {
    v8f acc = {0.f, 0.f, 0.f, 0.f, 0.f, 0.f, 0.f, 0.f};
    for (int k0 = 0; k0 < DH; k0 += 32) {
      v16bf a = load_a_rm(Q, DH, mBase, k0, lane);
      v16bf b = load_bt_rm(Km, DH, k0, nt * 16, lane);
      acc = wmma_bf16(a, b, acc);
    }
    int colb = nt * 16 + (lane & 15);
#pragma unroll
    for (int r = 0; r < 8; ++r) {
      int rr = r + ((lane & 16) ? 8 : 0);
      float s = acc[r] * 0.125f;
      if (colb >= TOPK) s = -1e30f;
      sc[rr][colb] = s;
    }
  }
  __syncthreads();

  {
    int row = threadIdx.x >> 3, sub = threadIdx.x & 7;
    float mx = -1e30f;
    for (int c = sub; c < KPAD; c += 8) mx = fmaxf(mx, sc[row][c]);
    for (int o = 4; o >= 1; o >>= 1) mx = fmaxf(mx, __shfl_xor(mx, o, 32));
    float sum = 0.f;
    for (int c = sub; c < KPAD; c += 8) {
      float e = __expf(sc[row][c] - mx);
      sc[row][c] = e;
      sum += e;
    }
    for (int o = 4; o >= 1; o >>= 1) sum += __shfl_xor(sum, o, 32);
    float inv = 1.0f / sum;
    for (int c = sub; c < KPAD; c += 8) at[row][c] = f2bf(sc[row][c] * inv);
  }
  __syncthreads();

  v8f acc = {0.f, 0.f, 0.f, 0.f, 0.f, 0.f, 0.f, 0.f};
  for (int k0 = 0; k0 < KPAD; k0 += 32) {
    v16bf a = load_a_rm(&at[0][0], KPAD, 0, k0, lane);
    v16bf b = load_b_rm(V, DH, k0, wid * 16, lane);
    acc = wmma_bf16(a, b, acc);
  }
  int b = bh / H, h = bh % H;
  int col = wid * 16 + (lane & 15);
#pragma unroll
  for (int r = 0; r < 8; ++r) {
    int kk = mBase + r + ((lane & 16) ? 8 : 0);
    ctx[((size_t)(b * KPAD + kk)) * D + h * DH + col] = f2bf(acc[r]);
  }
}

// Scatter attn output / passthrough raw tokens, store residual f32 + LN2 (bf16).
__global__ void scatter_ln2_kernel(const float* __restrict__ vis, const float* __restrict__ aud,
                                   const float* __restrict__ imu, const float* __restrict__ attn_out,
                                   const int* __restrict__ selpos, const float* __restrict__ g,
                                   const float* __restrict__ beta, float* __restrict__ resid,
                                   __bf16* __restrict__ hb) {
  __shared__ float red[256];
  int r = blockIdx.x;
  int b = r / T, t = r % T;
  int pos = selpos[b * T + t];
  const float* src;
  if (pos >= 0) {
    src = attn_out + ((size_t)(b * KPAD + pos)) * D;
  } else if (t < LV) {
    src = vis + ((size_t)b * LV + t) * D;
  } else if (t < LV + LA) {
    src = aud + ((size_t)b * LA + (t - LV)) * D;
  } else {
    src = imu + ((size_t)b * LI + (t - LV - LA)) * D;
  }
  float vals[4];
  float s = 0.f, sq = 0.f;
#pragma unroll
  for (int i = 0; i < 4; ++i) {
    int d = threadIdx.x + i * 256;
    float x = src[d];
    vals[i] = x;
    resid[(size_t)r * D + d] = x;
    s += x; sq += x * x;
  }
  s  = block_reduce_sum(s, red);
  sq = block_reduce_sum(sq, red);
  float m   = s * (1.0f / D);
  float var = sq * (1.0f / D) - m * m;
  float rs  = rsqrtf(var + 1e-5f);
#pragma unroll
  for (int i = 0; i < 4; ++i) {
    int d = threadIdx.x + i * 256;
    float y = (vals[i] - m) * rs * g[d] + beta[d];
    hb[(size_t)r * D + d] = f2bf(y);
  }
}

// ---------------- host orchestration ----------------
extern "C" void kernel_launch(void* const* d_in, const int* in_sizes, int n_in,
                              void* d_out, int out_size, void* d_ws, size_t ws_size,
                              hipStream_t stream) {
  (void)in_sizes; (void)n_in; (void)out_size; (void)ws_size;

  const float* visual = (const float*)d_in[0];
  const float* audio  = (const float*)d_in[1];
  const float* imu    = (const float*)d_in[2];
  const float* memb   = (const float*)d_in[3];
  const float* ln1_g  = (const float*)d_in[4];
  const float* ln1_b  = (const float*)d_in[5];
  const float* ln2_g  = (const float*)d_in[6];
  const float* ln2_b  = (const float*)d_in[7];
  const float* w_cur  = (const float*)d_in[8];
  const float* b_cur  = (const float*)d_in[9];
  const float* wq = (const float*)d_in[10]; const float* bq = (const float*)d_in[11];
  const float* wk = (const float*)d_in[12]; const float* bk = (const float*)d_in[13];
  const float* wv = (const float*)d_in[14]; const float* bv = (const float*)d_in[15];
  const float* wo = (const float*)d_in[16]; const float* bo = (const float*)d_in[17];
  const float* w1 = (const float*)d_in[18]; const float* b1 = (const float*)d_in[19];
  const float* w2 = (const float*)d_in[20]; const float* b2 = (const float*)d_in[21];

  float* out     = (float*)d_out;
  float* out_vis = out;
  float* out_aud = out_vis + (size_t)BB * LV * D;
  float* out_imu = out_aud + (size_t)BB * LA * D;
  float* out_mem = out_imu + (size_t)BB * LI * D;

  size_t off = 0;
  auto alloc = [&](size_t bytes) -> void* {
    off = (off + 255) & ~(size_t)255;
    void* p = (char*)d_ws + off;
    off += bytes;
    return p;
  };
  __bf16* wq_b  = (__bf16*)alloc((size_t)D * D * 2);
  __bf16* wk_b  = (__bf16*)alloc((size_t)D * D * 2);
  __bf16* wv_b  = (__bf16*)alloc((size_t)D * D * 2);
  __bf16* wo_b  = (__bf16*)alloc((size_t)D * D * 2);
  __bf16* w1_b  = (__bf16*)alloc((size_t)D * FF * 2);
  __bf16* w2_b  = (__bf16*)alloc((size_t)FF * D * 2);
  __bf16* xn_b  = (__bf16*)alloc((size_t)BB * T * D * 2);
  int*    idx   = (int*)   alloc((size_t)BB * KPAD * 4);
  int*    selp  = (int*)   alloc((size_t)BB * T * 4);
  __bf16* xsel  = (__bf16*)alloc((size_t)BB * KPAD * D * 2);
  __bf16* qh    = (__bf16*)alloc((size_t)BB * H * KPAD * DH * 2);
  __bf16* kh    = (__bf16*)alloc((size_t)BB * H * KPAD * DH * 2);
  __bf16* vh    = (__bf16*)alloc((size_t)BB * H * KPAD * DH * 2);
  __bf16* ctxb  = (__bf16*)alloc((size_t)BB * KPAD * D * 2);
  float*  aout  = (float*) alloc((size_t)BB * KPAD * D * 4);
  float*  resid = (float*) alloc((size_t)BB * T * D * 4);
  __bf16* hb    = (__bf16*)alloc((size_t)BB * T * D * 2);
  __bf16* ffh   = (__bf16*)alloc((size_t)BB * T * FF * 2);

  auto cvt = [&](const float* s, __bf16* d, long n) {
    cvt_bf16_kernel<<<(unsigned)((n + 255) / 256), 256, 0, stream>>>(s, d, n);
  };
  cvt(wq, wq_b, (long)D * D);
  cvt(wk, wk_b, (long)D * D);
  cvt(wv, wv_b, (long)D * D);
  cvt(wo, wo_b, (long)D * D);
  cvt(w1, w1_b, (long)D * FF);
  cvt(w2, w2_b, (long)FF * D);

  ln1_membrane_kernel<<<BB * T, 256, 0, stream>>>(visual, audio, imu, ln1_g, ln1_b,
                                                  w_cur, b_cur, memb, xn_b, out_mem);
  topk_kernel<<<BB, 1024, 0, stream>>>(out_mem, idx, selp);
  gather_kernel<<<BB * KPAD, 256, 0, stream>>>(xn_b, idx, xsel);

  // QKV projections
  {
    int grid = (BB * KPAD / BM) * (D / BN);   // 34 * 16
    gemm_tiled_kernel<0><<<grid, 256, 0, stream>>>(xsel, wq_b, bq, qh, nullptr, nullptr,
                                                   nullptr, nullptr, BB * KPAD, D, D);
    gemm_tiled_kernel<0><<<grid, 256, 0, stream>>>(xsel, wk_b, bk, kh, nullptr, nullptr,
                                                   nullptr, nullptr, BB * KPAD, D, D);
    gemm_tiled_kernel<0><<<grid, 256, 0, stream>>>(xsel, wv_b, bv, vh, nullptr, nullptr,
                                                   nullptr, nullptr, BB * KPAD, D, D);
  }

  attn_kernel<<<BB * H * MTILES, 128, 0, stream>>>(qh, kh, vh, ctxb);

  // O-projection
  {
    int grid = (BB * KPAD / BM) * (D / BN);
    gemm_tiled_kernel<1><<<grid, 256, 0, stream>>>(ctxb, wo_b, bo, nullptr, aout, nullptr,
                                                   nullptr, nullptr, BB * KPAD, D, D);
  }

  scatter_ln2_kernel<<<BB * T, 256, 0, stream>>>(visual, audio, imu, aout, selp,
                                                 ln2_g, ln2_b, resid, hb);

  // FFN GEMM1 + GELU
  {
    int grid = (BB * T / BM) * (FF / BN);     // 112 * 64
    gemm_tiled_kernel<2><<<grid, 256, 0, stream>>>(hb, w1_b, b1, ffh, nullptr, nullptr,
                                                   nullptr, nullptr, BB * T, FF, D);
  }

  // FFN GEMM2 + residual + split-write
  {
    int grid = (BB * T / BM) * (D / BN);      // 112 * 16
    gemm_tiled_kernel<3><<<grid, 256, 0, stream>>>(ffh, w2_b, b2, nullptr, out_vis, out_aud,
                                                   out_imu, resid, BB * T, D, FF);
  }
}
